// Attention_51874615001678
// MI455X (gfx1250) — compile-verified, matrix operand order checked by reading, not run on hardware
//
#include <hip/hip_runtime.h>
#include <hip/hip_bf16.h>

// ---------------------------------------------------------------------------
// Attention w/ stochastic-uncertainty mask for MI455X (gfx1250, wave32).
// All GEMMs via v_wmma_f32_16x16x32_bf16 (fp32 accumulate); softmax/tanh fp32.
// Pipeline is HBM-bound on the [B,H,N,N] tensors (~1.1 GB total traffic),
// so GEMM precision = bf16 tensor cores, and mask generation is fused into
// the score GEMM (all 12 heads' qk tiles held in registers for the 12x12
// cross-head mix).
// ---------------------------------------------------------------------------

typedef __bf16 bf16_t;
typedef __attribute__((ext_vector_type(16))) __bf16 v16bf;
typedef __attribute__((ext_vector_type(8)))  __bf16 v8bf;
typedef __attribute__((ext_vector_type(8)))  float  v8f;

static constexpr int Bb = 4;
static constexpr int Nn = 1024;
static constexpr int Cc = 768;
static constexpr int Hh = 12;
static constexpr int Dd = 64;     // head dim

#define WMMA_BF16(a, b, c) \
  __builtin_amdgcn_wmma_f32_16x16x32_bf16(false, (a), false, (b), (short)0, (c), false, false)

// Load a 16-lane-striped bf16 fragment (A-pattern; B uses the mirrored
// identical pattern for row-major [N,K] weight operands).
// rowptr = base of this lane's row; kbase already includes the +8 for the
// upper half-wave. Elements 0..7 = K[kbase..kbase+7], 8..15 = K[kbase+16..+23].
__device__ __forceinline__ v16bf load_frag(const bf16_t* __restrict__ rowptr, int kbase) {
  const v8bf lo = *(const v8bf*)(rowptr + kbase);
  const v8bf hi = *(const v8bf*)(rowptr + kbase + 16);
  v16bf f;
#pragma unroll
  for (int i = 0; i < 8; ++i) { f[i] = lo[i]; f[i + 8] = hi[i]; }
  return f;
}

// ---------------------------------------------------------------------------
// Kernel 0: f32 -> bf16 elementwise convert
// ---------------------------------------------------------------------------
__global__ void cvt_f32_to_bf16(const float* __restrict__ src, bf16_t* __restrict__ dst, int n) {
  const int i = blockIdx.x * blockDim.x + threadIdx.x;
  if (i < n) dst[i] = (bf16_t)src[i];
}

// ---------------------------------------------------------------------------
// Kernel 1: qkv = x @ qkv_w^T + qkv_b ; scatter to q,k (bf16 [B,H,N,hd]) and
// v transposed (bf16 [B,H,hd,N]).  One wave owns a 16x64 output tile.
// ---------------------------------------------------------------------------
__global__ void qkv_gemm(const bf16_t* __restrict__ xbf, const bf16_t* __restrict__ wbf,
                         const float* __restrict__ bias,
                         bf16_t* __restrict__ qb, bf16_t* __restrict__ kb,
                         bf16_t* __restrict__ vT) {
  const int lane  = threadIdx.x & 31;
  const int wid   = blockIdx.x * (blockDim.x >> 5) + (threadIdx.x >> 5);
  const int NT    = (3 * Cc) / 64;            // 36 col-tile groups
  const int tm    = wid / NT;                 // 0..255
  const int tn    = wid % NT;
  const int lrow  = lane & 15;
  const int lhalf = lane >> 4;
  const int kadd  = lhalf * 8;

  const bf16_t* arow = xbf + (size_t)(tm * 16 + lrow) * Cc;
  const bf16_t* wrow[4];
#pragma unroll
  for (int t = 0; t < 4; ++t)
    wrow[t] = wbf + (size_t)(tn * 64 + t * 16 + lrow) * Cc;

  v8f acc[4] = {};
  for (int kk = 0; kk < Cc; kk += 32) {
    const v16bf af = load_frag(arow, kk + kadd);
#pragma unroll
    for (int t = 0; t < 4; ++t) {
      const v16bf bfr = load_frag(wrow[t], kk + kadd);
      acc[t] = WMMA_BF16(af, bfr, acc[t]);
    }
  }

#pragma unroll
  for (int t = 0; t < 4; ++t) {
    const int c     = tn * 64 + t * 16 + lrow;   // column in [0, 2304)
    const int which = c / Cc;                    // 0=q, 1=k, 2=v
    const int rem   = c % Cc;
    const int h     = rem >> 6;
    const int d     = rem & 63;
    const float bs  = bias[c];
#pragma unroll
    for (int j = 0; j < 8; ++j) {
      const int mo  = tm * 16 + j + 8 * lhalf;   // global row (b*N + tok)
      const int bI  = mo >> 10;
      const int tok = mo & 1023;
      const bf16_t v = (bf16_t)(acc[t][j] + bs);
      if (which == 0)
        qb[((size_t)(bI * Hh + h) * Nn + tok) * Dd + d] = v;
      else if (which == 1)
        kb[((size_t)(bI * Hh + h) * Nn + tok) * Dd + d] = v;
      else
        vT[((size_t)(bI * Hh + h) * Dd + d) * Nn + tok] = v;
    }
  }
}

// ---------------------------------------------------------------------------
// Kernel 2: per 16x16 (i,j) tile: qk for all 12 heads (2 WMMAs each, K=64),
// then cross-head 1x1-conv mix + tanh + stochastic mask.  Raw scores go into
// the attn output buffer (scratch until softmax pass); mask -> byte buffer.
// grid = B * 64 * 16 blocks of 4 waves; wave w handles tj = tjg*4 + w.
// ---------------------------------------------------------------------------
__global__ void scores_mask(const bf16_t* __restrict__ qb, const bf16_t* __restrict__ kb,
                            const float* __restrict__ du_w, const float* __restrict__ du_b,
                            const float* __restrict__ r,
                            float* __restrict__ attnraw, unsigned char* __restrict__ maskbuf) {
  __shared__ float sduw[Hh * Hh];
  __shared__ float sdub[Hh];
  for (int i = threadIdx.x; i < Hh * Hh; i += blockDim.x) sduw[i] = du_w[i];
  if (threadIdx.x < Hh) sdub[threadIdx.x] = du_b[threadIdx.x];
  __syncthreads();

  const int lane  = threadIdx.x & 31;
  const int w     = threadIdx.x >> 5;
  int id          = blockIdx.x;
  const int tjg   = id & 15;  id >>= 4;
  const int ti    = id & 63;  id >>= 6;
  const int b     = id;
  const int tj    = tjg * 4 + w;
  const int lrow  = lane & 15;
  const int lhalf = lane >> 4;
  const int kadd  = lhalf * 8;

  v8f qk[Hh];
#pragma unroll
  for (int h = 0; h < Hh; ++h) {
    const bf16_t* qrow = qb + ((size_t)((b * Hh + h) * Nn) + ti * 16 + lrow) * Dd;
    const bf16_t* krow = kb + ((size_t)((b * Hh + h) * Nn) + tj * 16 + lrow) * Dd;
    v8f acc = {};
    acc = WMMA_BF16(load_frag(qrow, kadd),      load_frag(krow, kadd),      acc);
    acc = WMMA_BF16(load_frag(qrow, 32 + kadd), load_frag(krow, 32 + kadd), acc);
    qk[h] = acc;
  }

#pragma unroll
  for (int g = 0; g < Hh; ++g) {
    float u[8];
#pragma unroll
    for (int j = 0; j < 8; ++j) u[j] = sdub[g];
#pragma unroll
    for (int h = 0; h < Hh; ++h) {
      const float wgt = sduw[g * Hh + h];
#pragma unroll
      for (int j = 0; j < 8; ++j) u[j] = fmaf(wgt, qk[h][j], u[j]);
    }
    const size_t rowbase = (size_t)(b * Hh + g) * Nn;
#pragma unroll
    for (int j = 0; j < 8; ++j) {
      const int i  = ti * 16 + j + 8 * lhalf;
      const int jj = tj * 16 + lrow;
      const size_t idx = (rowbase + i) * (size_t)Nn + jj;
      const float unc  = 0.5f * (tanhf(u[j]) + 1.0f);
      maskbuf[idx] = (r[idx] > unc) ? (unsigned char)1 : (unsigned char)0;
      attnraw[idx] = qk[g][j];   // raw qk: softmax'd in next pass
    }
  }
}

// ---------------------------------------------------------------------------
// Kernel 3: in-place row softmax(qk * hd^-0.5) * mask.  One 256-thread block
// per (b,h,i) row of 1024 elements.
// ---------------------------------------------------------------------------
__global__ void softmax_mask(float* __restrict__ attn, const unsigned char* __restrict__ maskbuf) {
  const size_t row = blockIdx.x;
  float* p = attn + row * Nn;
  const unsigned char* mp = maskbuf + row * Nn;
  const int t = threadIdx.x;
  __shared__ float red[256];

  float x[4];
#pragma unroll
  for (int e = 0; e < 4; ++e) x[e] = p[t + e * 256];
  float mx = fmaxf(fmaxf(x[0], x[1]), fmaxf(x[2], x[3]));
  red[t] = mx;
  __syncthreads();
  for (int s = 128; s > 0; s >>= 1) { if (t < s) red[t] = fmaxf(red[t], red[t + s]); __syncthreads(); }
  mx = red[0];
  __syncthreads();

  const float scale = 0.125f;   // 64^-0.5
  float ex[4], sm = 0.0f;
#pragma unroll
  for (int e = 0; e < 4; ++e) { ex[e] = __expf(scale * (x[e] - mx)); sm += ex[e]; }
  red[t] = sm;
  __syncthreads();
  for (int s = 128; s > 0; s >>= 1) { if (t < s) red[t] += red[t + s]; __syncthreads(); }
  const float inv = 1.0f / red[0];

#pragma unroll
  for (int e = 0; e < 4; ++e)
    p[t + e * 256] = ex[e] * inv * (float)mp[t + e * 256];
}

// ---------------------------------------------------------------------------
// Kernel 4: ctx[b,i,h*64+d] = attn[b,h,i,:] @ v[b,h,:,d].  A (fp32 attn) is
// converted to bf16 fragments in registers; B = vT rows.  Wave owns 16x64.
// ---------------------------------------------------------------------------
__global__ void attn_v(const float* __restrict__ attn, const bf16_t* __restrict__ vT,
                       bf16_t* __restrict__ ctx) {
  const int lane  = threadIdx.x & 31;
  const int wid   = blockIdx.x * (blockDim.x >> 5) + (threadIdx.x >> 5);  // [0, B*H*64)
  const int ti    = wid & 63;
  const int bh    = wid >> 6;           // b*12 + h
  const int h     = bh % Hh;
  const int b     = bh / Hh;
  const int lrow  = lane & 15;
  const int lhalf = lane >> 4;
  const int kadd  = lhalf * 8;

  const float* arow = attn + ((size_t)(bh * Nn) + ti * 16 + lrow) * Nn;
  const bf16_t* vrow[4];
#pragma unroll
  for (int t = 0; t < 4; ++t)
    vrow[t] = vT + ((size_t)bh * Dd + t * 16 + lrow) * Nn;

  v8f acc[4] = {};
  for (int kk = 0; kk < Nn; kk += 32) {
    const int kb = kk + kadd;
    const float4 f0 = *(const float4*)(arow + kb);
    const float4 f1 = *(const float4*)(arow + kb + 4);
    const float4 f2 = *(const float4*)(arow + kb + 16);
    const float4 f3 = *(const float4*)(arow + kb + 20);
    v16bf af;
    af[0]  = (bf16_t)f0.x; af[1]  = (bf16_t)f0.y; af[2]  = (bf16_t)f0.z; af[3]  = (bf16_t)f0.w;
    af[4]  = (bf16_t)f1.x; af[5]  = (bf16_t)f1.y; af[6]  = (bf16_t)f1.z; af[7]  = (bf16_t)f1.w;
    af[8]  = (bf16_t)f2.x; af[9]  = (bf16_t)f2.y; af[10] = (bf16_t)f2.z; af[11] = (bf16_t)f2.w;
    af[12] = (bf16_t)f3.x; af[13] = (bf16_t)f3.y; af[14] = (bf16_t)f3.z; af[15] = (bf16_t)f3.w;
#pragma unroll
    for (int t = 0; t < 4; ++t)
      acc[t] = WMMA_BF16(af, load_frag(vrow[t], kb), acc[t]);
  }

#pragma unroll
  for (int t = 0; t < 4; ++t) {
#pragma unroll
    for (int j = 0; j < 8; ++j) {
      const int i = ti * 16 + j + 8 * lhalf;
      const int c = h * Dd + t * 16 + lrow;
      ctx[((size_t)b * Nn + i) * Cc + c] = (bf16_t)acc[t][j];
    }
  }
}

// ---------------------------------------------------------------------------
// Kernel 5: out = ctx @ proj_w^T + proj_b  (fp32 output).  Wave owns 16x64.
// ---------------------------------------------------------------------------
__global__ void proj_gemm(const bf16_t* __restrict__ ctx, const bf16_t* __restrict__ wbf,
                          const float* __restrict__ bias, float* __restrict__ out) {
  const int lane  = threadIdx.x & 31;
  const int wid   = blockIdx.x * (blockDim.x >> 5) + (threadIdx.x >> 5);
  const int NT    = Cc / 64;            // 12
  const int tm    = wid / NT;
  const int tn    = wid % NT;
  const int lrow  = lane & 15;
  const int lhalf = lane >> 4;
  const int kadd  = lhalf * 8;

  const bf16_t* arow = ctx + (size_t)(tm * 16 + lrow) * Cc;
  const bf16_t* wrow[4];
#pragma unroll
  for (int t = 0; t < 4; ++t)
    wrow[t] = wbf + (size_t)(tn * 64 + t * 16 + lrow) * Cc;

  v8f acc[4] = {};
  for (int kk = 0; kk < Cc; kk += 32) {
    const v16bf af = load_frag(arow, kk + kadd);
#pragma unroll
    for (int t = 0; t < 4; ++t)
      acc[t] = WMMA_BF16(af, load_frag(wrow[t], kk + kadd), acc[t]);
  }

#pragma unroll
  for (int t = 0; t < 4; ++t) {
    const int c    = tn * 64 + t * 16 + lrow;
    const float bs = bias[c];
#pragma unroll
    for (int j = 0; j < 8; ++j) {
      const int mo = tm * 16 + j + 8 * lhalf;
      out[(size_t)mo * Cc + c] = acc[t][j] + bs;
    }
  }
}

// ---------------------------------------------------------------------------
extern "C" void kernel_launch(void* const* d_in, const int* in_sizes, int n_in,
                              void* d_out, int out_size, void* d_ws, size_t ws_size,
                              hipStream_t stream) {
  const float* x      = (const float*)d_in[0];
  const float* r      = (const float*)d_in[1];
  const float* qkv_w  = (const float*)d_in[2];
  const float* qkv_b  = (const float*)d_in[3];
  const float* du_w   = (const float*)d_in[4];
  const float* du_b   = (const float*)d_in[5];
  const float* proj_w = (const float*)d_in[6];
  const float* proj_b = (const float*)d_in[7];

  float* out  = (float*)d_out;                       // [B,N,C]
  float* attn = out + (size_t)Bb * Nn * Cc;          // [B,H,N,N] (also raw-score scratch)

  // workspace layout (~86.5 MB total)
  char* ws = (char*)d_ws;
  size_t off = 0;
  auto carve = [&](size_t bytes) {
    void* p = ws + off;
    off += (bytes + 255) & ~(size_t)255;
    return p;
  };
  bf16_t* xbf    = (bf16_t*)carve((size_t)Bb * Nn * Cc * 2);
  bf16_t* qkvwb  = (bf16_t*)carve((size_t)3 * Cc * Cc * 2);
  bf16_t* projwb = (bf16_t*)carve((size_t)Cc * Cc * 2);
  bf16_t* qb     = (bf16_t*)carve((size_t)Bb * Hh * Nn * Dd * 2);
  bf16_t* kb     = (bf16_t*)carve((size_t)Bb * Hh * Nn * Dd * 2);
  bf16_t* vT     = (bf16_t*)carve((size_t)Bb * Hh * Dd * Nn * 2);
  bf16_t* ctx    = (bf16_t*)carve((size_t)Bb * Nn * Cc * 2);
  unsigned char* maskbuf = (unsigned char*)carve((size_t)Bb * Hh * Nn * Nn);

  const int nX = Bb * Nn * Cc, nQW = 3 * Cc * Cc, nPW = Cc * Cc;
  cvt_f32_to_bf16<<<(nX  + 255) / 256, 256, 0, stream>>>(x,      xbf,    nX);
  cvt_f32_to_bf16<<<(nQW + 255) / 256, 256, 0, stream>>>(qkv_w,  qkvwb,  nQW);
  cvt_f32_to_bf16<<<(nPW + 255) / 256, 256, 0, stream>>>(proj_w, projwb, nPW);

  // 256 M-tiles * 36 N-tiles = 9216 waves / 4 waves per block
  qkv_gemm<<<2304, 128, 0, stream>>>(xbf, qkvwb, qkv_b, qb, kb, vT);

  // B * 64 i-tiles * 16 j-tile-groups (4 waves each)
  scores_mask<<<Bb * 64 * 16, 128, 0, stream>>>(qb, kb, du_w, du_b, r, attn, maskbuf);

  softmax_mask<<<Bb * Hh * Nn, 256, 0, stream>>>(attn, maskbuf);

  // B*H*64 = 3072 waves / 4 per block
  attn_v<<<768, 128, 0, stream>>>(attn, vT, ctx);

  // 256 M-tiles * 12 N-tiles = 3072 waves / 4 per block
  proj_gemm<<<768, 128, 0, stream>>>(ctx, projwb, proj_b, out);
}
